// SpikeVAE_2070174236901
// MI455X (gfx1250) — matrix operand order repaired and checked by reading, not compiled, not run on hardware
//
#include <hip/hip_runtime.h>
#include <cstdint>

// Problem constants (match reference)
#define Bsz   128
#define NFc   64
#define Lw    4096
#define Kd    64
#define NGc   400
// linspace(0, 4096, 4096) -> step = 4096/4095 (endpoint inclusive)
#define STEPX (4096.0f / 4095.0f)
// exp(-(d^2)/(2*16^2)) = exp(d^2 * (-1/512))
#define NEGINV2S2 (-1.0f / 512.0f)

typedef __attribute__((ext_vector_type(16))) _Float16 v16h;
typedef __attribute__((ext_vector_type(8)))  _Float16 v8h;
typedef __attribute__((ext_vector_type(8)))  float    v8f;

#define LBLK 256      // L-chunk staged in LDS per iteration
#define ROWP 264      // padded LDS row stride (halves): 264*2B=528B=33*16B -> conflict-free b128

__device__ __forceinline__ float eluf(float x)  { return x > 0.0f ? x : expm1f(x); }
__device__ __forceinline__ float geluf(float x) { return 0.5f * x * (1.0f + erff(x * 0.70710678118654752f)); }

// Raw (pre-conversion) A fragment: 16 floats per lane
struct Araw { float4 a0, a1, a2, a3; };

__device__ __forceinline__ Araw load_rawA(const float* __restrict__ xbase, int off, int hi) {
    // A layout (16-bit A 16x32): lanes 0-15 hold K {0..7, 16..23}, lanes 16-31 K {8..15, 24..31}
    const float* xp = xbase + off + hi * 8;
    Araw r;
    r.a0 = *(const float4*)(xp +  0);
    r.a1 = *(const float4*)(xp +  4);
    r.a2 = *(const float4*)(xp + 16);
    r.a3 = *(const float4*)(xp + 20);
    return r;
}

__device__ __forceinline__ v16h cvtA(const Araw& r) {
    v16h A;
    A[0]=(_Float16)r.a0.x;  A[1]=(_Float16)r.a0.y;  A[2]=(_Float16)r.a0.z;  A[3]=(_Float16)r.a0.w;
    A[4]=(_Float16)r.a1.x;  A[5]=(_Float16)r.a1.y;  A[6]=(_Float16)r.a1.z;  A[7]=(_Float16)r.a1.w;
    A[8]=(_Float16)r.a2.x;  A[9]=(_Float16)r.a2.y;  A[10]=(_Float16)r.a2.z; A[11]=(_Float16)r.a2.w;
    A[12]=(_Float16)r.a3.x; A[13]=(_Float16)r.a3.y; A[14]=(_Float16)r.a3.z; A[15]=(_Float16)r.a3.w;
    return A;
}

// ---------------------------------------------------------------------------
// Kernel 0: zero the filtered-accumulator (128x64 f32) in workspace
// ---------------------------------------------------------------------------
__global__ void k_zero(float* __restrict__ p, int n) {
    int i = blockIdx.x * blockDim.x + threadIdx.x;
    if (i < n) p[i] = 0.0f;
}

// ---------------------------------------------------------------------------
// Kernel 1: filtered[b,f] += (1/L) * sum_l X[b,g,l] * amp[f,g]*exp(-(x_l-mu[f,g])^2/512)
// One workgroup per g. 8 waves x (16 rows of B) x (4 N-tiles of 16 filters).
// Filter slab generated on the fly into LDS; contraction via v_wmma_f32_16x16x32_f16.
// Software-pipelined: next A-fragment global loads issue before current WMMAs;
// the first load of the next L-block issues before the filter-gen barrier so
// the exp work hides its latency.
// ---------------------------------------------------------------------------
__global__ __launch_bounds__(256) void k_filter_gemm(
    const float* __restrict__ X,      // (B, NG, L)
    const float* __restrict__ mus,    // (NF, NG)
    const float* __restrict__ amps,   // (NF, NG)
    float* __restrict__ acc_out)      // (B, NF) f32 accumulator (pre-zeroed)
{
    __shared__ _Float16 sfilt[NFc * ROWP];   // [f][l_local] padded

    const int g    = blockIdx.x;
    const int t    = threadIdx.x;
    const int lane = t & 31;
    const int wave = t >> 5;        // 0..7 -> batch rows 16*wave..16*wave+15
    const int r    = lane & 15;
    const int hi   = lane >> 4;

    // Filter-generation assignment: thread t -> filter row gf, quarter gq (64 l's)
    const int   gf  = t >> 2;
    const int   gq  = t & 3;
    const float mu  = mus[gf * NGc + g];
    const float amp = amps[gf * NGc + g];

    v8f acc[4] = {v8f{}, v8f{}, v8f{}, v8f{}};

    const float* xbase = X + ((size_t)(wave * 16 + r) * NGc + g) * (size_t)Lw;

    // Pipeline prologue: first A fragment of the first block
    Araw raw = load_rawA(xbase, 0, hi);

    for (int kb = 0; kb < Lw; kb += LBLK) {
        __syncthreads();   // previous block's LDS reads complete
        {
            _Float16* dst = &sfilt[gf * ROWP + gq * 64];
            #pragma unroll
            for (int j8 = 0; j8 < 8; ++j8) {
                v8h pk;
                #pragma unroll
                for (int j = 0; j < 8; ++j) {
                    float xl = (float)(kb + gq * 64 + j8 * 8 + j) * STEPX;
                    float d  = xl - mu;
                    pk[j] = (_Float16)(amp * __expf(d * d * NEGINV2S2));
                }
                *(v8h*)(dst + j8 * 8) = pk;   // 16B aligned: ds_store_b128
            }
        }
        __syncthreads();

        // Convert the prefetched fragment (loads issued last step / before barrier)
        v16h A = cvtA(raw);

        #pragma unroll
        for (int s = 0; s < LBLK; s += 32) {
            // ---- Prefetch next A fragment (next step, or next block's step 0)
            const int nxt = kb + s + 32;
            if (nxt < Lw) raw = load_rawA(xbase, nxt, hi);

            // ---- Gather all 4 B fragments (8 ds_load_b128 in flight together).
            // lane column n = (lane&15); per-lane K = {hi*8 + j, 16 + hi*8 + j} (LDS [f][l])
            v16h Bf[4];
            #pragma unroll
            for (int nt = 0; nt < 4; ++nt) {
                const v8h* bp = (const v8h*)&sfilt[(nt * 16 + r) * ROWP + s + hi * 8];
                v8h b0 = bp[0], b1 = bp[2];
                #pragma unroll
                for (int j = 0; j < 8; ++j) { Bf[nt][j] = b0[j]; Bf[nt][8 + j] = b1[j]; }
            }

            // ---- 4 back-to-back WMMAs on the shared A fragment
            #pragma unroll
            for (int nt = 0; nt < 4; ++nt)
                acc[nt] = __builtin_amdgcn_wmma_f32_16x16x32_f16(
                    false, A, false, Bf[nt], (short)0, acc[nt], false, false);

            // ---- Convert prefetched raw for the next step (loadcnt wait lands here,
            // after the WMMAs). Last step of block: conversion deferred past barrier.
            if (s + 32 < LBLK) A = cvtA(raw);
        }
    }

    // ---- Flush partial tile. C layout: lane holds N=(lane&15); VGPR v -> M = hi*8 + v.
    const float scale = 1.0f / (float)Lw;
    #pragma unroll
    for (int nt = 0; nt < 4; ++nt) {
        const int f = nt * 16 + r;
        #pragma unroll
        for (int v = 0; v < 8; ++v) {
            const int b = wave * 16 + hi * 8 + v;
            atomicAdd(&acc_out[b * NFc + f], acc[nt][v] * scale);
        }
    }
}

// ---------------------------------------------------------------------------
// Kernel 2a: per batch row: mu, log_sig (elu), reparameterize, h = gelu(latent@W01^T+b01)
// ---------------------------------------------------------------------------
__global__ __launch_bounds__(64) void k_mlp1(
    const float* __restrict__ filtered, // (B, NF) from ws
    const float* __restrict__ eps,      // (B, K)
    const float* __restrict__ Wmu, const float* __restrict__ bmu,
    const float* __restrict__ Wsig, const float* __restrict__ bsig,
    const float* __restrict__ W01, const float* __restrict__ b01,
    float* __restrict__ out_mu, float* __restrict__ out_ls,
    float* __restrict__ h_out)          // (B, K) in ws
{
    __shared__ float fl[Kd];
    __shared__ float lat[Kd];
    const int b = blockIdx.x;
    const int t = threadIdx.x;   // output unit index

    fl[t] = filtered[b * NFc + t];
    __syncthreads();

    float s1 = bmu[t], s2 = bsig[t];
    #pragma unroll 8
    for (int f = 0; f < NFc; ++f) {
        const float v = fl[f];
        s1 = fmaf(v, Wmu[t * NFc + f], s1);
        s2 = fmaf(v, Wsig[t * NFc + f], s2);
    }
    const float m  = eluf(s1);
    const float ls = eluf(s2);
    out_mu[b * Kd + t] = m;
    out_ls[b * Kd + t] = ls;
    lat[t] = fmaf(eps[b * Kd + t], __expf(ls), m);
    __syncthreads();

    float s3 = b01[t];
    #pragma unroll 8
    for (int j = 0; j < Kd; ++j)
        s3 = fmaf(lat[j], W01[t * Kd + j], s3);
    h_out[b * Kd + t] = geluf(s3);
}

// ---------------------------------------------------------------------------
// Kernel 2b: coms = gelu(h@W1^T+b1), pred = gelu(h@W2^T+b2), both (B, NG)
// ---------------------------------------------------------------------------
__global__ __launch_bounds__(256) void k_mlp2(
    const float* __restrict__ h,   // (B, K)
    const float* __restrict__ W1, const float* __restrict__ b1,
    const float* __restrict__ W2, const float* __restrict__ b2,
    float* __restrict__ coms, float* __restrict__ pred)
{
    const int i = blockIdx.x * 256 + threadIdx.x;
    if (i >= Bsz * NGc) return;
    const int b = i / NGc;
    const int n = i - b * NGc;
    const float* hr = h + b * Kd;
    float s1 = b1[n], s2 = b2[n];
    #pragma unroll 8
    for (int k = 0; k < Kd; ++k) {
        const float hv = hr[k];
        s1 = fmaf(hv, W1[n * Kd + k], s1);
        s2 = fmaf(hv, W2[n * Kd + k], s2);
    }
    coms[i] = geluf(s1);
    pred[i] = geluf(s2);
}

// ---------------------------------------------------------------------------
extern "C" void kernel_launch(void* const* d_in, const int* in_sizes, int n_in,
                              void* d_out, int out_size, void* d_ws, size_t ws_size,
                              hipStream_t stream) {
    (void)in_sizes; (void)n_in; (void)out_size; (void)ws_size;
    const float* X    = (const float*)d_in[0];
    const float* eps  = (const float*)d_in[1];
    const float* mus  = (const float*)d_in[2];
    const float* amps = (const float*)d_in[3];
    const float* Wmu  = (const float*)d_in[4];
    const float* bmu  = (const float*)d_in[5];
    const float* Wsig = (const float*)d_in[6];
    const float* bsig = (const float*)d_in[7];
    const float* W01  = (const float*)d_in[8];
    const float* b01  = (const float*)d_in[9];
    const float* W1   = (const float*)d_in[10];
    const float* b1   = (const float*)d_in[11];
    const float* W2   = (const float*)d_in[12];
    const float* b2   = (const float*)d_in[13];

    float* out    = (float*)d_out;
    float* coms   = out;                       // (128,400)
    float* pred   = out + Bsz * NGc;           // (128,400)
    float* out_mu = out + 2 * Bsz * NGc;       // (128,64)
    float* out_ls = out + 2 * Bsz * NGc + Bsz * Kd;

    float* ws_filtered = (float*)d_ws;                 // 128*64 f32 = 32 KB
    float* ws_h        = ws_filtered + Bsz * NFc;      // 128*64 f32

    k_zero<<<(Bsz * NFc + 255) / 256, 256, 0, stream>>>(ws_filtered, Bsz * NFc);
    k_filter_gemm<<<NGc, 256, 0, stream>>>(X, mus, amps, ws_filtered);
    k_mlp1<<<Bsz, 64, 0, stream>>>(ws_filtered, eps, Wmu, bmu, Wsig, bsig, W01, b01,
                                   out_mu, out_ls, ws_h);
    k_mlp2<<<(Bsz * NGc + 255) / 256, 256, 0, stream>>>(ws_h, W1, b1, W2, b2, coms, pred);
}
